// TransformerEncoder_66494683677035
// MI455X (gfx1250) — matrix operand order repaired and checked by reading, not compiled, not run on hardware
//
#include <hip/hip_runtime.h>
#include <hip/hip_bf16.h>

// ---------------------------------------------------------------------------
// CDNA5 (gfx1250) transformer encoder forward.
// All GEMMs use v_wmma_f32_16x16x32_f16 (wave32 WMMA, f32 accumulate).
// Token GEMMs: 32x64 wave tile (8 WMMAs per K-step, A-frag amortization).
// ---------------------------------------------------------------------------

typedef __attribute__((ext_vector_type(16))) _Float16 v16h;
typedef __attribute__((ext_vector_type(8)))  _Float16 v8h;
typedef __attribute__((ext_vector_type(8)))  float    v8f;

static constexpr int Bb   = 4;
static constexpr int Ss   = 512;
static constexpr int Hh   = 768;
static constexpr int NHh  = 12;
static constexpr int Dd   = 64;
static constexpr int Ff   = 3072;
static constexpr int Ll   = 6;
static constexpr int Mtok = Bb * Ss;           // 2048 tokens

__device__ __forceinline__ v16h combine16(v8h lo, v8h hi) {
  return __builtin_shufflevector(lo, hi, 0, 1, 2, 3, 4, 5, 6, 7,
                                         8, 9, 10, 11, 12, 13, 14, 15);
}

// A fragment (16x32 f16): lane<16 -> K[k0..k0+7],[k0+16..k0+23]; lane>=16 -> +8
__device__ __forceinline__ v16h load_a_frag(const _Float16* __restrict__ pa) {
  v8h lo = *(const v8h*)(pa);
  v8h hi = *(const v8h*)(pa + 16);
  return combine16(lo, hi);
}
// B fragment from Bt[N,K]: lane<16 -> K[k0..k0+7],[k0+16..k0+23]; lane>=16 -> +8
__device__ __forceinline__ v16h load_b_frag(const _Float16* __restrict__ pb) {
  v8h lo = *(const v8h*)(pb);
  v8h hi = *(const v8h*)(pb + 16);
  return combine16(lo, hi);
}

// ---------------------------------------------------------------------------
// Token GEMM: C[M,N] = A[M,K] (f16, row major) x Bt[N,K] (f16) + bias
// Wave: 32 rows x 64 cols.  Block: 4 waves = 128x64 tile.
// ---------------------------------------------------------------------------
template <bool RELU, bool WF32, bool WF16, bool TRANSV>
__global__ __launch_bounds__(128)
void gemm_tok_kernel(const _Float16* __restrict__ A,
                     const _Float16* __restrict__ Bt,
                     const float* __restrict__ bias,
                     float* __restrict__ Cf,
                     _Float16* __restrict__ Ch,
                     int M, int N, int K) {
  const int lane  = threadIdx.x & 31;
  const int wave  = threadIdx.x >> 5;
  const int row0  = blockIdx.x * 128 + wave * 32;
  const int col0  = blockIdx.y * 64;
  const int mloc  = lane & 15;
  const int khalf = lane >> 4;

  v8f acc[2][4] = {};
  const _Float16* pa0 = A + (size_t)(row0 + mloc) * K + khalf * 8;
  const _Float16* pa1 = pa0 + (size_t)16 * K;

  for (int k0 = 0; k0 < K; k0 += 32) {
    v16h a0 = load_a_frag(pa0 + k0);
    v16h a1 = load_a_frag(pa1 + k0);
    if (k0 + 32 < K) {  // overlap next K-step's A rows with the WMMA stream
      __builtin_prefetch(pa0 + k0 + 32, 0, 1);
      __builtin_prefetch(pa1 + k0 + 32, 0, 1);
    }
#pragma unroll
    for (int t = 0; t < 4; ++t) {
      const _Float16* pb =
          Bt + (size_t)(col0 + t * 16 + mloc) * K + k0 + khalf * 8;
      v16h bfrag = load_b_frag(pb);
      acc[0][t] = __builtin_amdgcn_wmma_f32_16x16x32_f16(
          false, a0, false, bfrag, (short)0, acc[0][t], false, false);
      acc[1][t] = __builtin_amdgcn_wmma_f32_16x16x32_f16(
          false, a1, false, bfrag, (short)0, acc[1][t], false, false);
    }
  }

#pragma unroll
  for (int half = 0; half < 2; ++half) {
#pragma unroll
    for (int t = 0; t < 4; ++t) {
      const int gc = col0 + t * 16 + mloc;
      const float bv = bias[gc];
#pragma unroll
      for (int r = 0; r < 8; ++r) {
        const int gr = row0 + half * 16 + khalf * 8 + r;
        float v = acc[half][t][r] + bv;
        if (RELU) v = v > 0.f ? v : 0.f;
        if (WF32) Cf[(size_t)gr * N + gc] = v;
        if (WF16) {
          if (TRANSV) {  // store V transposed: Vt[b, col(H), s]
            const int b = gr >> 9, s = gr & 511;
            Ch[((size_t)b * Hh + gc) * Ss + s] = (_Float16)v;
          } else {
            Ch[(size_t)gr * N + gc] = (_Float16)v;
          }
        }
      }
    }
  }
}

// ---------------------------------------------------------------------------
// Attention scores: logits[b,h,q,k] = scale * Q_bh[q,:] . K_bh[k,:]
// Q,K stored [B*S, H] f16; head slice = 64 cols at h*64, row stride H.
// ---------------------------------------------------------------------------
__global__ __launch_bounds__(128)
void score_gemm_kernel(const _Float16* __restrict__ Q,
                       const _Float16* __restrict__ Kmat,
                       float* __restrict__ logits, float scale) {
  const int lane  = threadIdx.x & 31;
  const int wave  = threadIdx.x >> 5;
  const int bh    = blockIdx.z;
  const int b     = bh / NHh;
  const int h     = bh % NHh;
  const int row0  = blockIdx.x * 64 + wave * 16;
  const int col0  = blockIdx.y * 64;
  const int mloc  = lane & 15;
  const int khalf = lane >> 4;

  const _Float16* Abase = Q    + (size_t)b * Ss * Hh + h * Dd;
  const _Float16* Bbase = Kmat + (size_t)b * Ss * Hh + h * Dd;

  v8f acc[4] = {};
  const _Float16* pa0 = Abase + (size_t)(row0 + mloc) * Hh + khalf * 8;
  for (int k0 = 0; k0 < Dd; k0 += 32) {
    v16h afrag = load_a_frag(pa0 + k0);
#pragma unroll
    for (int t = 0; t < 4; ++t) {
      const _Float16* pb =
          Bbase + (size_t)(col0 + t * 16 + mloc) * Hh + k0 + khalf * 8;
      v16h bfrag = load_b_frag(pb);
      acc[t] = __builtin_amdgcn_wmma_f32_16x16x32_f16(
          false, afrag, false, bfrag, (short)0, acc[t], false, false);
    }
  }

  float* out = logits + (size_t)bh * Ss * Ss;
#pragma unroll
  for (int t = 0; t < 4; ++t) {
    const int gc = col0 + t * 16 + mloc;
#pragma unroll
    for (int r = 0; r < 8; ++r) {
      const int gr = row0 + khalf * 8 + r;
      out[(size_t)gr * Ss + gc] = acc[t][r] * scale;
    }
  }
}

// ---------------------------------------------------------------------------
// ctx = P @ V : P f16 [bh, S, S], Vt f16 [b, H, S]  ->  ctx f16 [B*S, H]
// ---------------------------------------------------------------------------
__global__ __launch_bounds__(128)
void pv_gemm_kernel(const _Float16* __restrict__ P,
                    const _Float16* __restrict__ Vt,
                    _Float16* __restrict__ Ctx) {
  const int lane  = threadIdx.x & 31;
  const int wave  = threadIdx.x >> 5;
  const int bh    = blockIdx.z;
  const int b     = bh / NHh;
  const int h     = bh % NHh;
  const int row0  = blockIdx.x * 64 + wave * 16;
  const int mloc  = lane & 15;
  const int khalf = lane >> 4;

  const _Float16* Abase = P  + (size_t)bh * Ss * Ss;
  const _Float16* Bbase = Vt + ((size_t)b * Hh + h * Dd) * Ss;

  v8f acc[4] = {};
  const _Float16* pa0 = Abase + (size_t)(row0 + mloc) * Ss + khalf * 8;
  for (int k0 = 0; k0 < Ss; k0 += 32) {
    v16h afrag = load_a_frag(pa0 + k0);
#pragma unroll
    for (int t = 0; t < 4; ++t) {
      const _Float16* pb =
          Bbase + (size_t)(t * 16 + mloc) * Ss + k0 + khalf * 8;
      v16h bfrag = load_b_frag(pb);
      acc[t] = __builtin_amdgcn_wmma_f32_16x16x32_f16(
          false, afrag, false, bfrag, (short)0, acc[t], false, false);
    }
  }

#pragma unroll
  for (int t = 0; t < 4; ++t) {
    const int gc = t * 16 + mloc;
#pragma unroll
    for (int r = 0; r < 8; ++r) {
      const int gr = row0 + khalf * 8 + r;
      Ctx[((size_t)b * Ss + gr) * Hh + h * Dd + gc] = (_Float16)acc[t][r];
    }
  }
}

// ---------------------------------------------------------------------------
// Row softmax with padding mask; in-place on f32 logits, f16 copy for PV GEMM.
// One block (256 threads) per row of 512.
// ---------------------------------------------------------------------------
__global__ __launch_bounds__(256)
void softmax_kernel(float* __restrict__ logits, const int* __restrict__ tok,
                    _Float16* __restrict__ Ph) {
  const int idx = blockIdx.x;       // bh*512 + q
  const int q   = idx & 511;
  const int bh  = idx >> 9;
  const int b   = bh / NHh;
  float* row = logits + (size_t)bh * Ss * Ss + (size_t)q * Ss;
  _Float16* prow = Ph + (size_t)bh * Ss * Ss + (size_t)q * Ss;

  const int t = threadIdx.x;
  float v0 = row[t]       + (tok[b * Ss + t]       == 0 ? -10000.f : 0.f);
  float v1 = row[t + 256] + (tok[b * Ss + t + 256] == 0 ? -10000.f : 0.f);

  __shared__ float red[256];
  red[t] = fmaxf(v0, v1);
  __syncthreads();
  for (int s = 128; s > 0; s >>= 1) {
    if (t < s) red[t] = fmaxf(red[t], red[t + s]);
    __syncthreads();
  }
  const float m = red[0];
  __syncthreads();

  float e0 = __expf(v0 - m), e1 = __expf(v1 - m);
  red[t] = e0 + e1;
  __syncthreads();
  for (int s = 128; s > 0; s >>= 1) {
    if (t < s) red[t] += red[t + s];
    __syncthreads();
  }
  const float inv = 1.f / red[0];

  const float p0 = e0 * inv, p1 = e1 * inv;
  row[t] = p0;
  row[t + 256] = p1;
  prow[t] = (_Float16)p0;
  prow[t + 256] = (_Float16)p1;
}

// ---------------------------------------------------------------------------
// x = LayerNorm(x + y) * g + b ; also emit f16 copy. One block per token row.
// ---------------------------------------------------------------------------
__global__ __launch_bounds__(256)
void add_ln_kernel(const float* __restrict__ X, const float* __restrict__ Y,
                   const float* __restrict__ g, const float* __restrict__ be,
                   float* __restrict__ Xo, _Float16* __restrict__ Xh) {
  const int row = blockIdx.x;
  const int t = threadIdx.x;
  const float* xr = X + (size_t)row * Hh;
  const float* yr = Y + (size_t)row * Hh;

  float v[3];
  float s = 0.f;
#pragma unroll
  for (int i = 0; i < 3; ++i) {
    v[i] = xr[t + i * 256] + yr[t + i * 256];
    s += v[i];
  }
  __shared__ float red[256];
  red[t] = s;
  __syncthreads();
  for (int k = 128; k > 0; k >>= 1) {
    if (t < k) red[t] += red[t + k];
    __syncthreads();
  }
  const float mean = red[0] * (1.f / (float)Hh);
  __syncthreads();

  float var = 0.f;
#pragma unroll
  for (int i = 0; i < 3; ++i) {
    const float d = v[i] - mean;
    var += d * d;
  }
  red[t] = var;
  __syncthreads();
  for (int k = 128; k > 0; k >>= 1) {
    if (t < k) red[t] += red[t + k];
    __syncthreads();
  }
  const float rstd = rsqrtf(red[0] * (1.f / (float)Hh) + 1e-12f);

#pragma unroll
  for (int i = 0; i < 3; ++i) {
    const int c = t + i * 256;
    const float o = (v[i] - mean) * rstd * g[c] + be[c];
    Xo[(size_t)row * Hh + c] = o;
    Xh[(size_t)row * Hh + c] = (_Float16)o;
  }
}

// ---------------------------------------------------------------------------
// Embedding gather + sinusoidal pos add; emits f32 state and f16 copy.
// ---------------------------------------------------------------------------
__global__ __launch_bounds__(256)
void embed_kernel(const int* __restrict__ tok, const float* __restrict__ emb,
                  const float* __restrict__ pos, float* __restrict__ X,
                  _Float16* __restrict__ Xh) {
  const int row = blockIdx.x;       // 0..2047
  const int s = row & 511;
  const int t = threadIdx.x;
  const int id = tok[row];
#pragma unroll
  for (int i = 0; i < 3; ++i) {
    const int c = t + i * 256;
    const float v = emb[(size_t)id * Hh + c] + pos[(size_t)s * Hh + c];
    X[(size_t)row * Hh + c] = v;
    Xh[(size_t)row * Hh + c] = (_Float16)v;
  }
}

// ---------------------------------------------------------------------------
// W f32 [K,N] row-major  ->  Wt f16 [N,K]  (transpose + convert)
// ---------------------------------------------------------------------------
__global__ __launch_bounds__(256)
void transpose_cvt_kernel(const float* __restrict__ W,
                          _Float16* __restrict__ Wt, int K, int N) {
  const size_t i = (size_t)blockIdx.x * 256 + threadIdx.x;
  const size_t total = (size_t)K * N;
  if (i >= total) return;
  const size_t n = i / (size_t)K;
  const size_t k = i % (size_t)K;
  Wt[i] = (_Float16)W[k * (size_t)N + n];
}

// ---------------------------------------------------------------------------
// Host side
// ---------------------------------------------------------------------------
extern "C" void kernel_launch(void* const* d_in, const int* in_sizes, int n_in,
                              void* d_out, int out_size, void* d_ws,
                              size_t ws_size, hipStream_t stream) {
  const int*   tok   = (const int*)d_in[0];
  const float* emb   = (const float*)d_in[1];
  const float* pos   = (const float*)d_in[2];
  const float* wq    = (const float*)d_in[3];
  const float* bq    = (const float*)d_in[4];
  const float* wk    = (const float*)d_in[5];
  const float* bk    = (const float*)d_in[6];
  const float* wv    = (const float*)d_in[7];
  const float* bv    = (const float*)d_in[8];
  const float* wo    = (const float*)d_in[9];
  const float* bo    = (const float*)d_in[10];
  const float* ln1g  = (const float*)d_in[11];
  const float* ln1b  = (const float*)d_in[12];
  const float* w1    = (const float*)d_in[13];
  const float* b1    = (const float*)d_in[14];
  const float* w2    = (const float*)d_in[15];
  const float* b2    = (const float*)d_in[16];
  const float* ln2g  = (const float*)d_in[17];
  const float* ln2b  = (const float*)d_in[18];

  char* ws = (char*)d_ws;
  // Static scratch layout (bytes, 256-aligned)
  float*    xf32  = (float*)(ws + 0);            // 2048*768 f32
  _Float16* xh    = (_Float16*)(ws + 6291456);   // 2048*768 f16
  _Float16* qh    = (_Float16*)(ws + 9437184);
  _Float16* kh    = (_Float16*)(ws + 12582912);
  _Float16* vt    = (_Float16*)(ws + 15728640);  // [B,H,S] transposed V
  _Float16* ctxh  = (_Float16*)(ws + 18874368);
  _Float16* a1h   = (_Float16*)(ws + 22020096);  // 2048*3072 f16
  float*    yf32  = (float*)(ws + 34603008);     // 2048*768 f32
  _Float16* attnh = (_Float16*)(ws + 40894464);  // B*NH*S*S f16
  _Float16* wtq   = (_Float16*)(ws + 66060288);  // per-layer reused
  _Float16* wtk   = (_Float16*)(ws + 67239936);
  _Float16* wtv   = (_Float16*)(ws + 68419584);
  _Float16* wto   = (_Float16*)(ws + 69599232);
  _Float16* wt1   = (_Float16*)(ws + 70778880);  // [F,H]
  _Float16* wt2   = (_Float16*)(ws + 75497472);  // [H,F]

  float* outp = (float*)d_out;
  float* attn_out = outp + (size_t)Mtok * Hh;    // [L,B,NH,S,S]

  const dim3 blk(128);
  const dim3 gHH(Mtok / 128, Hh / 64);           // 16 x 12
  const dim3 gHF(Mtok / 128, Ff / 64);           // 16 x 48
  const float scale = 0.125f;                    // 1/sqrt(64)

  embed_kernel<<<Mtok, 256, 0, stream>>>(tok, emb, pos, xf32, xh);

  for (int l = 0; l < Ll; ++l) {
    const size_t oHH = (size_t)l * Hh * Hh;
    transpose_cvt_kernel<<<(Hh * Hh + 255) / 256, 256, 0, stream>>>(wq + oHH, wtq, Hh, Hh);
    transpose_cvt_kernel<<<(Hh * Hh + 255) / 256, 256, 0, stream>>>(wk + oHH, wtk, Hh, Hh);
    transpose_cvt_kernel<<<(Hh * Hh + 255) / 256, 256, 0, stream>>>(wv + oHH, wtv, Hh, Hh);
    transpose_cvt_kernel<<<(Hh * Hh + 255) / 256, 256, 0, stream>>>(wo + oHH, wto, Hh, Hh);
    transpose_cvt_kernel<<<(Hh * Ff + 255) / 256, 256, 0, stream>>>(w1 + (size_t)l * Hh * Ff, wt1, Hh, Ff);
    transpose_cvt_kernel<<<(Hh * Ff + 255) / 256, 256, 0, stream>>>(w2 + (size_t)l * Ff * Hh, wt2, Ff, Hh);

    // Q, K, V projections (V stored transposed per head)
    gemm_tok_kernel<false, false, true, false><<<gHH, blk, 0, stream>>>(
        xh, wtq, bq + (size_t)l * Hh, nullptr, qh, Mtok, Hh, Hh);
    gemm_tok_kernel<false, false, true, false><<<gHH, blk, 0, stream>>>(
        xh, wtk, bk + (size_t)l * Hh, nullptr, kh, Mtok, Hh, Hh);
    gemm_tok_kernel<false, false, true, true><<<gHH, blk, 0, stream>>>(
        xh, wtv, bv + (size_t)l * Hh, nullptr, vt, Mtok, Hh, Hh);

    // Attention
    float* lgt = attn_out + (size_t)l * Bb * NHh * Ss * Ss;
    score_gemm_kernel<<<dim3(Ss / 64, Ss / 64, Bb * NHh), blk, 0, stream>>>(
        qh, kh, lgt, scale);
    softmax_kernel<<<Bb * NHh * Ss, 256, 0, stream>>>(lgt, tok, attnh);
    pv_gemm_kernel<<<dim3(Ss / 64, 1, Bb * NHh), blk, 0, stream>>>(
        attnh, vt, ctxh);

    // Output projection + LN1
    gemm_tok_kernel<false, true, false, false><<<gHH, blk, 0, stream>>>(
        ctxh, wto, bo + (size_t)l * Hh, yf32, nullptr, Mtok, Hh, Hh);
    add_ln_kernel<<<Mtok, 256, 0, stream>>>(
        xf32, yf32, ln1g + (size_t)l * Hh, ln1b + (size_t)l * Hh, xf32, xh);

    // FFN + LN2
    gemm_tok_kernel<true, false, true, false><<<gHF, blk, 0, stream>>>(
        xh, wt1, b1 + (size_t)l * Ff, nullptr, a1h, Mtok, Ff, Hh);
    gemm_tok_kernel<false, true, false, false><<<gHH, blk, 0, stream>>>(
        a1h, wt2, b2 + (size_t)l * Hh, yf32, nullptr, Mtok, Hh, Ff);
    add_ln_kernel<<<Mtok, 256, 0, stream>>>(
        xf32, yf32, ln2g + (size_t)l * Hh, ln2b + (size_t)l * Hh, xf32, xh);
  }

  hipMemcpyAsync(d_out, xf32, (size_t)Mtok * Hh * sizeof(float),
                 hipMemcpyDeviceToDevice, stream);
}
